// SimpleHGTModel_37855841747404
// MI455X (gfx1250) — matrix operand order repaired and checked by reading, not compiled, not run on hardware
//
#include <hip/hip_runtime.h>
#include <hip/hip_bf16.h>
#include <cstdint>

#define ND_ 10000
#define NG_ 20000
#define E_  300000
#define F_  128
#define H_  4
#define HF_ 512
#define DRUG_DIM_ 1127
#define GENE_DIM_ 4402

typedef __attribute__((ext_vector_type(16))) __bf16 v16bf;
typedef __attribute__((ext_vector_type(8)))  float  v8f;

// ---------------- helpers ----------------
__device__ __forceinline__ uint32_t f32_to_bf16_rne(float f) {
  uint32_t x = __float_as_uint(f);
  uint32_t r = x + 0x7fffu + ((x >> 16) & 1u);
  return r >> 16;
}
__device__ __forceinline__ uint32_t pack_bf16(float lo, float hi) {
  return f32_to_bf16_rne(lo) | (f32_to_bf16_rne(hi) << 16);
}
__device__ __forceinline__ uint32_t enc_ord(float f) {
  uint32_t u = __float_as_uint(f);
  return (u & 0x80000000u) ? ~u : (u | 0x80000000u);
}
__device__ __forceinline__ float dec_ord(uint32_t u) {
  u = (u & 0x80000000u) ? (u ^ 0x80000000u) : ~u;
  return __uint_as_float(u);
}
__device__ __forceinline__ float lrelu02(float x) { return x > 0.f ? x : 0.2f * x; }

// ---------------- WMMA bf16 GEMM: C[M,N] = act(A[M,K] @ B[K,N] + bias) ----------------
// Tile: 128(M) x 64(N), K-step 32. 256 threads = 8 waves; wave (wy 0..3, wx 0..1)
// owns a 32x32 quadrant = 2x2 subtiles of 16x16.
#define TM 128
#define TN 64
#define TK 32
#define ASTR 36   // lds_a row stride in halfwords (18 dwords; 8B aligned rows)
#define BSTR 68   // lds_b row stride in dwords (16B aligned rows)

union AF { v16bf v; uint32_t u[8]; };

__global__ __launch_bounds__(256)
void gemm_bf16_wmma(const float* __restrict__ A, const float* __restrict__ B,
                    const float* __restrict__ bias, float* __restrict__ C,
                    int M, int N, int K, int relu) {
  __shared__ uint16_t lds_a[TM * ASTR];          // bf16 bits
  __shared__ uint32_t lds_b[(TK / 2) * BSTR];    // packed {B[2k,n],B[2k+1,n]}

  const int tid  = threadIdx.x;
  const int wave = tid >> 5;
  const int lane = tid & 31;
  const int wy = wave & 3;      // M quadrant (0..3)
  const int wx = wave >> 2;     // N quadrant (0..1)
  const int half = lane >> 4;
  const int l15  = lane & 15;

  const int m0 = blockIdx.y * TM;
  const int n0 = blockIdx.x * TN;

  const bool mfull = (m0 + TM) <= M;
  const bool nfull = (n0 + TN) <= N;
  const bool k4    = (K & 3) == 0;

  v8f acc[2][2];
  #pragma unroll
  for (int s = 0; s < 2; ++s)
    #pragma unroll
    for (int t = 0; t < 2; ++t)
      acc[s][t] = v8f{0.f, 0.f, 0.f, 0.f, 0.f, 0.f, 0.f, 0.f};

  const uint32_t* la32 = (const uint32_t*)lds_a;

  for (int k0 = 0; k0 < K; k0 += TK) {
    const bool kfull = (k0 + TK) <= K;

    // ---- stage A (fp32 -> bf16) ----
    if (mfull && kfull && k4) {
      // vectorized: 1024 float4 chunks, 4 per thread; b128 load + b64 LDS store
      #pragma unroll
      for (int j = 0; j < 4; ++j) {
        int i = tid + j * 256;      // chunk id
        int m = i >> 3;             // 8 chunks (of 4) per 32-wide row
        int kq = (i & 7) * 4;
        const float4 v = *(const float4*)(A + (size_t)(m0 + m) * K + k0 + kq);
        uint2 p;
        p.x = pack_bf16(v.x, v.y);
        p.y = pack_bf16(v.z, v.w);
        *(uint2*)&lds_a[m * ASTR + kq] = p;
      }
    } else {
      #pragma unroll
      for (int i = tid; i < TM * TK; i += 256) {
        int m = i >> 5, k = i & 31;
        int gm = m0 + m, gk = k0 + k;
        float v = (gm < M && gk < K) ? A[(size_t)gm * K + gk] : 0.f;
        lds_a[m * ASTR + k] = (uint16_t)f32_to_bf16_rne(v);
      }
    }
    // ---- stage B (pairs along K packed into dwords) ----
    if (nfull && kfull) {
      // vectorized: 256 quads, 1 per thread; 2x b128 loads + b128 LDS store
      int kp = tid >> 4;            // 0..15
      int nq = (tid & 15) * 4;      // 0,4,...,60
      int gk = k0 + 2 * kp;
      const float4 r0 = *(const float4*)(B + (size_t)gk * N + n0 + nq);
      const float4 r1 = *(const float4*)(B + (size_t)(gk + 1) * N + n0 + nq);
      uint4 p;
      p.x = pack_bf16(r0.x, r1.x);
      p.y = pack_bf16(r0.y, r1.y);
      p.z = pack_bf16(r0.z, r1.z);
      p.w = pack_bf16(r0.w, r1.w);
      *(uint4*)&lds_b[kp * BSTR + nq] = p;
    } else {
      #pragma unroll
      for (int i = tid; i < (TK / 2) * TN; i += 256) {
        int kp = i >> 6, n = i & 63;
        int gk = k0 + 2 * kp, gn = n0 + n;
        float f0 = (gk < K && gn < N) ? B[(size_t)gk * N + gn] : 0.f;
        float f1 = ((gk + 1) < K && gn < N) ? B[(size_t)(gk + 1) * N + gn] : 0.f;
        lds_b[kp * BSTR + n] = pack_bf16(f0, f1);
      }
    }
    __syncthreads();

    AF af[2], bf[2];
    #pragma unroll
    for (int s = 0; s < 2; ++s) {
      int m = wy * 32 + s * 16 + l15;
      #pragma unroll
      for (int v = 0; v < 8; ++v) {
        int kp = ((v & 4) ? 16 : 0) + half * 8 + (v & 3) * 2; // A layout ISA 7.12.2
        af[s].u[v] = la32[m * (ASTR / 2) + (kp >> 1)];
      }
    }
    #pragma unroll
    for (int t = 0; t < 2; ++t) {
      int n = wx * 32 + t * 16 + l15;
      #pragma unroll
      for (int v = 0; v < 8; ++v) {
        int kp = ((v & 4) ? 8 : 0) + half * 4 + (v & 3);      // B: K pair matches A layout
        bf[t].u[v] = lds_b[kp * BSTR + n];
      }
    }
    #pragma unroll
    for (int s = 0; s < 2; ++s)
      #pragma unroll
      for (int t = 0; t < 2; ++t)
        acc[s][t] = __builtin_amdgcn_wmma_f32_16x16x32_bf16(
            false, af[s].v, false, bf[t].v, (short)0, acc[s][t], false, false);
    __syncthreads();
  }

  // ---- epilogue: C layout (ISA 7.12.2 32-bit C/D 16x16) ----
  #pragma unroll
  for (int s = 0; s < 2; ++s) {
    #pragma unroll
    for (int t = 0; t < 2; ++t) {
      int colg = n0 + wx * 32 + t * 16 + l15;
      #pragma unroll
      for (int r = 0; r < 8; ++r) {
        int rowg = m0 + wy * 32 + s * 16 + half * 8 + r;
        if (rowg < M && colg < N) {
          float v = acc[s][t][r] + (bias ? bias[colg] : 0.f);
          if (relu) v = fmaxf(v, 0.f);
          C[(size_t)rowg * N + colg] = v;
        }
      }
    }
  }
}

// ---------------- a[n,h] = sum_f h[n,h,f] * att[h,f] ----------------
__global__ __launch_bounds__(128)
void att_reduce(const float* __restrict__ hmat, const float* __restrict__ att,
                float* __restrict__ out, int Nn) {
  int n = blockIdx.x;
  if (n >= Nn) return;
  int h = threadIdx.x >> 5;
  int l = threadIdx.x & 31;
  const float* row = hmat + (size_t)n * HF_ + h * F_;
  const float* a   = att + h * F_;
  float sum = 0.f;
  #pragma unroll
  for (int j = 0; j < 4; ++j) sum += row[l + 32 * j] * a[l + 32 * j];
  #pragma unroll
  for (int off = 16; off; off >>= 1) sum += __shfl_down(sum, off, 32);
  if (l == 0) out[n * H_ + h] = sum;
}

// ---------------- edge pass 1: per-dst max of leaky_relu logits ----------------
__global__ __launch_bounds__(256)
void edge_max(const int* __restrict__ src, const int* __restrict__ dst,
              const float* __restrict__ as_, const float* __restrict__ ad_,
              uint32_t* __restrict__ menc, int nE) {
  int e = blockIdx.x * blockDim.x + threadIdx.x;
  if (e >= nE) return;
  int s = src[e], d = dst[e];
  #pragma unroll
  for (int h = 0; h < H_; ++h) {
    float lg = lrelu02(as_[s * H_ + h] + ad_[d * H_ + h]);
    atomicMax(&menc[d * H_ + h], enc_ord(lg));
  }
}

// ---------------- edge pass 2: per-dst sum of exp(logit - max) ----------------
__global__ __launch_bounds__(256)
void edge_expsum(const int* __restrict__ src, const int* __restrict__ dst,
                 const float* __restrict__ as_, const float* __restrict__ ad_,
                 const uint32_t* __restrict__ menc, float* __restrict__ ssum, int nE) {
  int e = blockIdx.x * blockDim.x + threadIdx.x;
  if (e >= nE) return;
  int s = src[e], d = dst[e];
  #pragma unroll
  for (int h = 0; h < H_; ++h) {
    float lg = lrelu02(as_[s * H_ + h] + ad_[d * H_ + h]);
    float m  = dec_ord(menc[d * H_ + h]);
    atomicAdd(&ssum[d * H_ + h], __expf(lg - m));
  }
}

// ---------------- edge pass 3: accum[dst] += alpha * hs[src] ----------------
__global__ __launch_bounds__(128)
void edge_scatter(const int* __restrict__ src, const int* __restrict__ dst,
                  const float* __restrict__ as_, const float* __restrict__ ad_,
                  const uint32_t* __restrict__ menc, const float* __restrict__ ssum,
                  const float* __restrict__ hsrc, float* __restrict__ accum, int nE) {
  __shared__ float alpha[H_];
  int e = blockIdx.x;
  if (e >= nE) return;
  int s = src[e], d = dst[e];
  int t = threadIdx.x;
  if (t < H_) {
    float lg = lrelu02(as_[s * H_ + t] + ad_[d * H_ + t]);
    float m  = dec_ord(menc[d * H_ + t]);
    alpha[t] = __expf(lg - m) / (ssum[d * H_ + t] + 1e-16f);
  }
  __syncthreads();
  const float* hrow = hsrc + (size_t)s * HF_;
  float* arow = accum + (size_t)d * HF_;
  #pragma unroll
  for (int h = 0; h < H_; ++h)
    atomicAdd(&arow[h * F_ + t], alpha[h] * hrow[h * F_ + t]);
}

// ---------------- out[n,f] = relu(mean_h accum[n,h,f] + b[f]) ----------------
__global__ __launch_bounds__(256)
void finalize_mean(const float* __restrict__ accum, const float* __restrict__ b,
                   float* __restrict__ out, int Nn) {
  int i = blockIdx.x * blockDim.x + threadIdx.x;
  if (i >= Nn * F_) return;
  int n = i >> 7, f = i & 127;
  const float* a = accum + (size_t)n * HF_;
  float v = 0.25f * (a[f] + a[F_ + f] + a[2 * F_ + f] + a[3 * F_ + f]) + b[f];
  out[i] = fmaxf(v, 0.f);
}

// ---------------- out[n] = dot(h[n,:], W2) + b2 ----------------
__global__ __launch_bounds__(256)
void mlp_out_gemv(const float* __restrict__ h, const float* __restrict__ W2,
                  const float* __restrict__ b2, float* __restrict__ out, int Nn) {
  int n = blockIdx.x * 8 + (threadIdx.x >> 5);
  int l = threadIdx.x & 31;
  if (n >= Nn) return;
  const float* row = h + (size_t)n * F_;
  float sum = 0.f;
  #pragma unroll
  for (int j = 0; j < 4; ++j) sum += row[l + 32 * j] * W2[l + 32 * j];
  #pragma unroll
  for (int off = 16; off; off >>= 1) sum += __shfl_down(sum, off, 32);
  if (l == 0) out[n] = sum + b2[0];
}

// ---------------- host orchestration ----------------
extern "C" void kernel_launch(void* const* d_in, const int* in_sizes, int n_in,
                              void* d_out, int out_size, void* d_ws, size_t ws_size,
                              hipStream_t stream) {
  (void)in_sizes; (void)n_in; (void)out_size; (void)ws_size;
  const float* x_drug = (const float*)d_in[0];
  const float* x_gene = (const float*)d_in[1];
  const int*   e_src  = (const int*)d_in[2];
  const int*   e_dst  = (const int*)d_in[3];
  const float* pdW = (const float*)d_in[4]; const float* pdb = (const float*)d_in[5];
  const float* pgW = (const float*)d_in[6]; const float* pgb = (const float*)d_in[7];
  const float* mlp_W1 = (const float*)d_in[28]; const float* mlp_b1 = (const float*)d_in[29];
  const float* mlp_W2 = (const float*)d_in[30]; const float* mlp_b2 = (const float*)d_in[31];

  // workspace carve-out
  char* ws = (char*)d_ws;
  size_t off = 0;
  auto alloc = [&](size_t bytes) -> void* {
    off = (off + 255) & ~(size_t)255;
    void* p = ws + off;
    off += bytes;
    return p;
  };
  float* xd0 = (float*)alloc((size_t)ND_ * F_ * 4);
  float* xg0 = (float*)alloc((size_t)NG_ * F_ * 4);
  float* xd1 = (float*)alloc((size_t)ND_ * F_ * 4);
  float* xg1 = (float*)alloc((size_t)NG_ * F_ * 4);
  float* hs  = (float*)alloc((size_t)NG_ * HF_ * 4);
  float* hd  = (float*)alloc((size_t)NG_ * HF_ * 4);
  float* acc = (float*)alloc((size_t)NG_ * HF_ * 4);
  float* a_s = (float*)alloc((size_t)NG_ * H_ * 4);
  float* a_d = (float*)alloc((size_t)NG_ * H_ * 4);
  uint32_t* menc = (uint32_t*)alloc((size_t)NG_ * H_ * 4);
  float* ssum = (float*)alloc((size_t)NG_ * H_ * 4);

  auto gemm = [&](const float* A, const float* B, const float* bias, float* C,
                  int M, int N, int K, int relu) {
    dim3 g((N + TN - 1) / TN, (M + TM - 1) / TM);
    gemm_bf16_wmma<<<g, 256, 0, stream>>>(A, B, bias, C, M, N, K, relu);
  };

  // input projections
  gemm(x_drug, pdW, pdb, xd0, ND_, F_, DRUG_DIM_, 0);
  gemm(x_gene, pgW, pgb, xg0, NG_, F_, GENE_DIM_, 0);

  float* xd = xd0; float* xg = xg0; float* xdn = xd1; float* xgn = xg1;
  const int egrid = (E_ + 255) / 256;

  for (int l = 0; l < 2; ++l) {
    int base = 8 + l * 10;
    // ---------- direction drug->gene (dst = genes) ----------
    {
      const float* Ws  = (const float*)d_in[base + 0];
      const float* Wd  = (const float*)d_in[base + 1];
      const float* ats = (const float*)d_in[base + 2];
      const float* atd = (const float*)d_in[base + 3];
      const float* b   = (const float*)d_in[base + 4];
      gemm(xd, Ws, nullptr, hs, ND_, HF_, F_, 0);
      gemm(xg, Wd, nullptr, hd, NG_, HF_, F_, 0);
      att_reduce<<<ND_, 128, 0, stream>>>(hs, ats, a_s, ND_);
      att_reduce<<<NG_, 128, 0, stream>>>(hd, atd, a_d, NG_);
      hipMemsetAsync(menc, 0, (size_t)NG_ * H_ * 4, stream);
      hipMemsetAsync(ssum, 0, (size_t)NG_ * H_ * 4, stream);
      hipMemsetAsync(acc, 0, (size_t)NG_ * HF_ * 4, stream);
      edge_max<<<egrid, 256, 0, stream>>>(e_src, e_dst, a_s, a_d, menc, E_);
      edge_expsum<<<egrid, 256, 0, stream>>>(e_src, e_dst, a_s, a_d, menc, ssum, E_);
      edge_scatter<<<E_, 128, 0, stream>>>(e_src, e_dst, a_s, a_d, menc, ssum, hs, acc, E_);
      finalize_mean<<<(NG_ * F_ + 255) / 256, 256, 0, stream>>>(acc, b, xgn, NG_);
    }
    // ---------- direction gene->drug (dst = drugs) ----------
    {
      const float* Ws  = (const float*)d_in[base + 5];
      const float* Wd  = (const float*)d_in[base + 6];
      const float* ats = (const float*)d_in[base + 7];
      const float* atd = (const float*)d_in[base + 8];
      const float* b   = (const float*)d_in[base + 9];
      gemm(xg, Ws, nullptr, hs, NG_, HF_, F_, 0);
      gemm(xd, Wd, nullptr, hd, ND_, HF_, F_, 0);
      att_reduce<<<NG_, 128, 0, stream>>>(hs, ats, a_s, NG_);
      att_reduce<<<ND_, 128, 0, stream>>>(hd, atd, a_d, ND_);
      hipMemsetAsync(menc, 0, (size_t)ND_ * H_ * 4, stream);
      hipMemsetAsync(ssum, 0, (size_t)ND_ * H_ * 4, stream);
      hipMemsetAsync(acc, 0, (size_t)ND_ * HF_ * 4, stream);
      // reversed edge roles: src = gene ids (e_dst), dst = drug ids (e_src)
      edge_max<<<egrid, 256, 0, stream>>>(e_dst, e_src, a_s, a_d, menc, E_);
      edge_expsum<<<egrid, 256, 0, stream>>>(e_dst, e_src, a_s, a_d, menc, ssum, E_);
      edge_scatter<<<E_, 128, 0, stream>>>(e_dst, e_src, a_s, a_d, menc, ssum, hs, acc, E_);
      finalize_mean<<<(ND_ * F_ + 255) / 256, 256, 0, stream>>>(acc, b, xdn, ND_);
    }
    // swap ping-pong
    float* t;
    t = xd; xd = xdn; xdn = t;
    t = xg; xg = xgn; xgn = t;
  }

  // final MLP: h = relu(xd @ W1 + b1); out = h @ W2 + b2
  float* hbuf = hs; // reuse
  gemm(xd, mlp_W1, mlp_b1, hbuf, ND_, F_, F_, 1);
  mlp_out_gemv<<<(ND_ + 7) / 8, 256, 0, stream>>>(hbuf, mlp_W2, mlp_b2, (float*)d_out, ND_);
}
